// LastBipartiteGraphMultiHeadAttentionLayer_12541304504945
// MI455X (gfx1250) — matrix operand order repaired and checked by reading, not compiled, not run on hardware
//
#include <hip/hip_runtime.h>
#include <hip/hip_bf16.h>

typedef __attribute__((ext_vector_type(16))) __bf16 v16bf;
typedef __attribute__((ext_vector_type(8)))  float  v8f;

#define NN      8192
#define MM      8192
#define FD      128
#define HEADS   4
#define DD      64
#define CHUNK   32
#define NCHUNK  (MM / CHUNK)     /* 256 */
#define ROWS    32               /* output rows per block in main kernel */

/* ---------------- Phase 1a: wa[h][f] = Wf[h][f][:] . a[h][:]  (4x128) -------- */
__global__ void k_wa(const float* __restrict__ Wf, const float* __restrict__ a,
                     float* __restrict__ wa) {
    int idx = blockIdx.x * blockDim.x + threadIdx.x;   // 0..511
    int h = idx >> 7, f = idx & 127;
    const float* w  = Wf + (((h << 7) + f) * DD);
    const float* av = a + h * DD;
    float s = 0.f;
#pragma unroll 8
    for (int d = 0; d < DD; ++d) s += w[d] * av[d];
    wa[idx] = s;
}

/* ---------------- Phase 1b: e_src[h][n] = F0[n][:] . wa[h][:] ---------------- */
__global__ void k_esrc(const float* __restrict__ F0, const float* __restrict__ wa,
                       float* __restrict__ e_src) {
    int lane = threadIdx.x & 31;
    int n = blockIdx.x * 8 + (threadIdx.x >> 5);       // one wave per row n
    const float4* row = (const float4*)(F0 + (long)n * FD);
    float4 fv = row[lane];                             // f = 4*lane .. 4*lane+3
    const float4* waq = (const float4*)wa;
#pragma unroll
    for (int h = 0; h < HEADS; ++h) {
        float4 wv = waq[h * 32 + lane];
        float p = fv.x * wv.x + fv.y * wv.y + fv.z * wv.z + fv.w * wv.w;
#pragma unroll
        for (int off = 16; off; off >>= 1) p += __shfl_xor(p, off, 32);
        if (lane == 0) e_src[h * NN + n] = p;
    }
}

/* ---------------- Phase 1c: E1 projection -----------------------------------
 * Per (h,m): E1[h][m][d] = E0[m][:] . We[h][:][d]   (64 threads, d = tid)
 * Emits:
 *   pk[h][m]  = {t = e_dst, exp(t), exp(0.2 t), 0}           (float4)
 *   E1b       = E1 in bf16, pre-swizzled into WMMA B-operand tile layout:
 *     tile (h, kc=m/32, dt=d/16) of 512 bf16; within tile:
 *     lane = (d&15) + (K&8?16:0), K=m&31, vgpr j=((K&7)>>1)+(K&16?4:0), half=K&1
 */
__global__ void k_proj_e(const float* __restrict__ E0, const float* __restrict__ We,
                         const float* __restrict__ a,
                         float4* __restrict__ pk, __hip_bfloat16* __restrict__ E1b) {
    int bid = blockIdx.x;
    int h = bid >> 13, m = bid & (MM - 1);
    int d = threadIdx.x;                              // 0..63
    const float* e0 = E0 + (long)m * FD;
    const float* w  = We + ((long)(h << 7)) * DD + d; // We[h][f][d], stride DD
    float v = 0.f;
#pragma unroll 8
    for (int f = 0; f < FD; ++f) v += e0[f] * w[f * DD];

    /* e_dst reduction across 64 threads (2 waves) */
    float p = v * a[h * DD + d];
#pragma unroll
    for (int off = 16; off; off >>= 1) p += __shfl_xor(p, off, 32);
    __shared__ float sp[2];
    if ((threadIdx.x & 31) == 0) sp[threadIdx.x >> 5] = p;
    __syncthreads();
    if (threadIdx.x == 0) {
        float t = sp[0] + sp[1];
        pk[h * MM + m] = make_float4(t, expf(t), expf(0.2f * t), 0.f);
    }

    /* bf16 swizzled store */
    int kc = m >> 5, K = m & 31;
    int dt = d >> 4;
    int lane = (d & 15) + ((K & 8) ? 16 : 0);
    int j = ((K & 7) >> 1) + ((K & 16) ? 4 : 0), half = K & 1;
    int idx = ((((h << 8) + kc) * 4 + dt) << 9) + lane * 16 + j * 2 + half;
    E1b[idx] = __float2bfloat16(v);
}

/* ---------------- Phase 2: masked softmax-attention with bf16 WMMA ---------- */
__global__ __launch_bounds__(256) void k_gat(
        const int* __restrict__ A, const float* __restrict__ e_src,
        const float4* __restrict__ pk, const uint4* __restrict__ gB,
        float* __restrict__ out) {
    __shared__ __align__(16) float sAf[2][ROWS][36];   /* A mask as float, padded */
    __shared__ __align__(16) float sT [2][HEADS][32];  /* t  = e_dst              */
    __shared__ __align__(16) float sE1[2][HEADS][32];  /* exp(t)                  */
    __shared__ __align__(16) float sE2[2][HEADS][32];  /* exp(0.2 t)              */
    __shared__ float  sOutH[2][HEADS][16][68];         /* padded                  */

    const int tid  = threadIdx.x;
    const int lane = tid & 31;
    const int wave = tid >> 5;
    const int head = wave & 3;
    const int nsub = wave >> 2;
    const int l15  = lane & 15;
    const int hi   = (lane & 16) ? 1 : 0;
    const int b    = hi ? 8 : 0;                       /* A-operand K base        */
    const int arow = nsub * 16 + l15;
    const int n    = blockIdx.x * ROWS + arow;

    const float s   = e_src[head * NN + n];
    const float es1 = expf(s);
    const float es2 = expf(0.2f * s);
    const float thr = -s;                              /* s + t > 0  <=>  t > -s */

    v8f acc0 = {0,0,0,0,0,0,0,0}, acc1 = {0,0,0,0,0,0,0,0};
    v8f acc2 = {0,0,0,0,0,0,0,0}, acc3 = {0,0,0,0,0,0,0,0};
    v8f accD = {0,0,0,0,0,0,0,0};                      /* row-sum denominators    */

    /* constant ones B-matrix (bf16 1.0 pairs) for the denominator WMMA */
    union { unsigned u[8]; v16bf v; } onesb;
#pragma unroll
    for (int q = 0; q < 8; ++q) onesb.u[q] = 0x3F803F80u;

    /* cooperative staging indices: 8 threads per A row, 4 ints each */
    const int strow = tid >> 3;
    const int stk   = (tid & 7) * 4;
    const long aBase = (long)(blockIdx.x * ROWS + strow) * MM;
    const int4* Aq = (const int4*)A;
    const int ph = tid >> 5, pkk = tid & 31;           /* pk staging (tid<128)    */

    /* preload chunk 0 */
    {
        int4 av = Aq[(aBase + stk) >> 2];
        *(float4*)&sAf[0][strow][stk] =
            make_float4((float)av.x, (float)av.y, (float)av.z, (float)av.w);
        if (tid < 128) {
            float4 pv = pk[ph * MM + pkk];
            sT [0][ph][pkk] = pv.x;
            sE1[0][ph][pkk] = pv.y;
            sE2[0][ph][pkk] = pv.z;
        }
    }
    __syncthreads();

    for (int c = 0; c < NCHUNK; ++c) {
        const int buf = c & 1;
        /* prefetch next chunk into registers */
        int4 nA; float4 nPK;
        const bool has = (c + 1 < NCHUNK);
        if (has) {
            nA = Aq[(aBase + (c + 1) * CHUNK + stk) >> 2];
            if (tid < 128) nPK = pk[ph * MM + (c + 1) * CHUNK + pkk];
        }
        if (c + 2 < NCHUNK)
            __builtin_prefetch(&Aq[(aBase + (c + 2) * CHUNK + stk) >> 2], 0, 1);

        /* build bf16 A-operand (16x32): p = Af * (t > -s ? es1*et1 : es2*et2) */
        unsigned au[8];
#pragma unroll
        for (int run = 0; run < 2; ++run) {
            const int kb = run * 16 + b;
            const float* tv = &sT [buf][head][kb];
            const float* e1 = &sE1[buf][head][kb];
            const float* e2 = &sE2[buf][head][kb];
            float4 t0  = *(const float4*)(tv);     float4 t1  = *(const float4*)(tv + 4);
            float4 x0  = *(const float4*)(e1);     float4 x1  = *(const float4*)(e1 + 4);
            float4 y0  = *(const float4*)(e2);     float4 y1  = *(const float4*)(e2 + 4);
            float4 m0  = *(const float4*)&sAf[buf][arow][kb];
            float4 m1  = *(const float4*)&sAf[buf][arow][kb + 4];
            const float* tp = (const float*)&t0;   /* t0,t1 contiguous */
            const float* xp = (const float*)&x0;
            const float* yp = (const float*)&y0;
            const float* mp = (const float*)&m0;
            float tt[8], xx[8], yy[8], mm[8];
#pragma unroll
            for (int i = 0; i < 4; ++i) {
                tt[i] = ((const float*)&t0)[i]; tt[i+4] = ((const float*)&t1)[i];
                xx[i] = ((const float*)&x0)[i]; xx[i+4] = ((const float*)&x1)[i];
                yy[i] = ((const float*)&y0)[i]; yy[i+4] = ((const float*)&y1)[i];
                mm[i] = ((const float*)&m0)[i]; mm[i+4] = ((const float*)&m1)[i];
            }
            (void)tp; (void)xp; (void)yp; (void)mp;
            float vals[8];
#pragma unroll
            for (int i = 0; i < 8; ++i) {
                bool cond = tt[i] > thr;
                float f = cond ? es1 : es2;
                float g = cond ? xx[i] : yy[i];
                vals[i] = (f * g) * mm[i];
            }
#pragma unroll
            for (int q = 0; q < 4; ++q) {
                au[run * 4 + q] = __builtin_amdgcn_perm(
                    __float_as_uint(vals[2 * q + 1]),
                    __float_as_uint(vals[2 * q]),
                    0x07060302u);                 /* {hi.b3,hi.b2,lo.b3,lo.b2}   */
            }
        }
        union { unsigned u[8]; v16bf v; } aop;
#pragma unroll
        for (int q = 0; q < 8; ++q) aop.u[q] = au[q];

        /* denominator row-sums: P (16x32) x Ones (32x16) */
        accD = __builtin_amdgcn_wmma_f32_16x16x32_bf16(false, aop.v, false, onesb.v, (short)0, accD, false, false);

        /* B tiles straight from L2 (pre-swizzled); 2 x b128 per 32x16 tile */
        const int tb = ((head * NCHUNK + c) * 4) * 64 + lane * 2;
        union { uint4 q[2]; v16bf v; } bop;
        bop.q[0] = gB[tb +   0]; bop.q[1] = gB[tb +   1];
        acc0 = __builtin_amdgcn_wmma_f32_16x16x32_bf16(false, aop.v, false, bop.v, (short)0, acc0, false, false);
        bop.q[0] = gB[tb +  64]; bop.q[1] = gB[tb +  65];
        acc1 = __builtin_amdgcn_wmma_f32_16x16x32_bf16(false, aop.v, false, bop.v, (short)0, acc1, false, false);
        bop.q[0] = gB[tb + 128]; bop.q[1] = gB[tb + 129];
        acc2 = __builtin_amdgcn_wmma_f32_16x16x32_bf16(false, aop.v, false, bop.v, (short)0, acc2, false, false);
        bop.q[0] = gB[tb + 192]; bop.q[1] = gB[tb + 193];
        acc3 = __builtin_amdgcn_wmma_f32_16x16x32_bf16(false, aop.v, false, bop.v, (short)0, acc3, false, false);

        __syncthreads();
        if (has) {
            *(float4*)&sAf[buf ^ 1][strow][stk] =
                make_float4((float)nA.x, (float)nA.y, (float)nA.z, (float)nA.w);
            if (tid < 128) {
                sT [buf ^ 1][ph][pkk] = nPK.x;
                sE1[buf ^ 1][ph][pkk] = nPK.y;
                sE2[buf ^ 1][ph][pkk] = nPK.z;
            }
        }
        __syncthreads();
    }

    /* accD[j] already holds the denominator of row (j + 8*hi) for every lane */
    const int mhi = hi ? 8 : 0;
    float rsc[8];
#pragma unroll
    for (int j = 0; j < 8; ++j)
        rsc[j] = 0.25f / fmaxf(accD[j], 1e-30f);       /* 0.25 = mean over heads */

    /* scatter per-head scaled tile to LDS (deterministic cross-head reduce) */
#pragma unroll
    for (int j = 0; j < 8; ++j) {
        sOutH[nsub][head][j + mhi][ 0 + l15] = acc0[j] * rsc[j];
        sOutH[nsub][head][j + mhi][16 + l15] = acc1[j] * rsc[j];
        sOutH[nsub][head][j + mhi][32 + l15] = acc2[j] * rsc[j];
        sOutH[nsub][head][j + mhi][48 + l15] = acc3[j] * rsc[j];
    }
    __syncthreads();

    /* cross-head sum + row softmax over d=64: 8 threads x 8 cols per row */
    const int r  = tid >> 3;
    const int cg = (tid & 7) * 8;
    const int ns = r >> 4, mr = r & 15;
    float v[8];
#pragma unroll
    for (int i = 0; i < 8; ++i)
        v[i] = sOutH[ns][0][mr][cg + i] + sOutH[ns][1][mr][cg + i]
             + sOutH[ns][2][mr][cg + i] + sOutH[ns][3][mr][cg + i];
    float mx = v[0];
#pragma unroll
    for (int i = 1; i < 8; ++i) mx = fmaxf(mx, v[i]);
#pragma unroll
    for (int off = 4; off; off >>= 1) mx = fmaxf(mx, __shfl_xor(mx, off, 32));
    float sum = 0.f;
#pragma unroll
    for (int i = 0; i < 8; ++i) { v[i] = expf(v[i] - mx); sum += v[i]; }
#pragma unroll
    for (int off = 4; off; off >>= 1) sum += __shfl_xor(sum, off, 32);
    float inv = 1.f / sum;
    float* po = out + (long)(blockIdx.x * ROWS + r) * DD + cg;
#pragma unroll
    for (int i = 0; i < 8; ++i) po[i] = v[i] * inv;
}

/* ---------------------------------------------------------------------------- */
extern "C" void kernel_launch(void* const* d_in, const int* in_sizes, int n_in,
                              void* d_out, int out_size, void* d_ws, size_t ws_size,
                              hipStream_t stream) {
    const float* F0 = (const float*)d_in[0];
    const float* E0 = (const float*)d_in[1];
    const int*   A  = (const int*)d_in[2];
    const float* Wf = (const float*)d_in[3];
    const float* We = (const float*)d_in[4];
    const float* a  = (const float*)d_in[5];
    float* out = (float*)d_out;

    char* ws = (char*)d_ws;
    float*          wa    = (float*)(ws);                               /* 2 KB   */
    float*          e_src = (float*)(ws + 2048);                        /* 128 KB */
    float4*         pk    = (float4*)(ws + 2048 + 131072);              /* 512 KB */
    __hip_bfloat16* E1b   = (__hip_bfloat16*)(ws + 2048 + 131072 + 524288); /* 4 MB */

    k_wa    <<<2,            256, 0, stream>>>(Wf, a, wa);
    k_esrc  <<<NN / 8,       256, 0, stream>>>(F0, wa, e_src);
    k_proj_e<<<HEADS * MM,    64, 0, stream>>>(E0, We, a, pk, E1b);
    k_gat   <<<NN / ROWS,    256, 0, stream>>>(A, e_src, pk, (const uint4*)E1b, out);
}
